// MultiHeadCrossAttention_88012469830219
// MI455X (gfx1250) — matrix operand order repaired and checked by reading, not compiled, not run on hardware
//
#include <hip/hip_runtime.h>

#define B_    32
#define S_    512
#define H_    12
#define HD_   64
#define HID_  768
#define IND_  512

typedef __attribute__((ext_vector_type(16))) __bf16 bf16x16;
typedef __attribute__((ext_vector_type(8)))  __bf16 bf16x8;
typedef __attribute__((ext_vector_type(4)))  __bf16 bf16x4;
typedef __attribute__((ext_vector_type(8)))  float  f32x8;

// ---------------------------------------------------------------------------
// WMMA helper: D = A(16x32 bf16) x B(32x16 bf16) + C(16x16 f32)
// ---------------------------------------------------------------------------
static __device__ inline f32x8 wmma_bf16(bf16x16 a, bf16x16 b, f32x8 c) {
  return __builtin_amdgcn_wmma_f32_16x16x32_bf16(false, a, false, b,
                                                 (short)0, c, false, false);
}

// Assemble a fragment from two 8-element contiguous bf16 runs (global or LDS).
static __device__ inline bf16x16 frag2x8(const __bf16* p0, const __bf16* p1) {
  bf16x8 a = *(const bf16x8*)p0;
  bf16x8 b = *(const bf16x8*)p1;
  bf16x16 r;
#pragma unroll
  for (int e = 0; e < 8; ++e) { r[e] = a[e]; r[8 + e] = b[e]; }
  return r;
}
// A-fragment (16x32, 16-bit): caller passes pointer already offset to
// (row, kc + khalf*16); the two 8-element runs are at +0 and +8.
static __device__ inline bf16x16 frag_a16(const __bf16* p) { return frag2x8(p, p + 8); }
// B-fragment (32x16, 16-bit): 16 contiguous values along K.
static __device__ inline bf16x16 frag_b16(const __bf16* p) { return frag2x8(p, p + 8); }

static __device__ inline float redmax16(float v) {
  v = fmaxf(v, __shfl_xor(v, 1));
  v = fmaxf(v, __shfl_xor(v, 2));
  v = fmaxf(v, __shfl_xor(v, 4));
  v = fmaxf(v, __shfl_xor(v, 8));
  return v;
}
static __device__ inline float redadd16(float v) {
  v += __shfl_xor(v, 1);
  v += __shfl_xor(v, 2);
  v += __shfl_xor(v, 4);
  v += __shfl_xor(v, 8);
  return v;
}
static __device__ inline float redadd32(float v) {
  v += __shfl_xor(v, 1);
  v += __shfl_xor(v, 2);
  v += __shfl_xor(v, 4);
  v += __shfl_xor(v, 8);
  v += __shfl_xor(v, 16);
  return v;
}

// ---------------------------------------------------------------------------
// Kernel 0: one-shot f32 -> bf16 conversion (4 elements / thread).
// ---------------------------------------------------------------------------
__global__ __launch_bounds__(256) void cvt_bf16_kernel(
    const float* __restrict__ src, __bf16* __restrict__ dst) {
  const size_t i = ((size_t)blockIdx.x * 256 + threadIdx.x) * 4;
  float4 x = *(const float4*)(src + i);
  bf16x4 y;
  y[0] = (__bf16)x.x; y[1] = (__bf16)x.y;
  y[2] = (__bf16)x.z; y[3] = (__bf16)x.w;
  *(bf16x4*)(dst + i) = y;
}

// ---------------------------------------------------------------------------
// Kernel 1: Y = l2norm(LN(X @ W^T + b)); X:[16384,512] bf16, W:[768,512] bf16
// One block = 16 rows x 768 cols. 8 waves x 6 N-tiles, K looped in 32-chunks.
// Emits f32 copy (residual / exact LN) and bf16 copy (attention A-operands).
// ---------------------------------------------------------------------------
__global__ __launch_bounds__(256) void proj_norm_kernel(
    const __bf16* __restrict__ img_xh, const __bf16* __restrict__ txt_xh,
    const __bf16* __restrict__ img_Wh, const float* __restrict__ img_b,
    const float* __restrict__ img_g, const float* __restrict__ img_beta,
    const __bf16* __restrict__ txt_Wh, const float* __restrict__ txt_b,
    const float* __restrict__ txt_g, const float* __restrict__ txt_beta,
    float* __restrict__ imgN, float* __restrict__ txtN,
    __bf16* __restrict__ imgNh, __bf16* __restrict__ txtNh) {
  __shared__ float rowbuf[16 * HID_];  // 48 KB

  const int tid = threadIdx.x, wave = tid >> 5, lane = tid & 31;
  const int n = lane & 15, khalf = lane >> 4;
  const int row0 = blockIdx.x * 16;
  const bool isTxt = (blockIdx.y != 0);

  const __bf16* X  = isTxt ? txt_xh   : img_xh;
  const __bf16* W  = isTxt ? txt_Wh   : img_Wh;
  const float*  bb = isTxt ? txt_b    : img_b;
  const float*  g  = isTxt ? txt_g    : img_g;
  const float*  be = isTxt ? txt_beta : img_beta;
  float*        Y  = isTxt ? txtN     : imgN;
  __bf16*       Yh = isTxt ? txtNh    : imgNh;

  f32x8 acc[6];
#pragma unroll
  for (int j = 0; j < 6; ++j) acc[j] = {};

  const int arow = row0 + n;
  for (int kc = 0; kc < IND_; kc += 32) {
    bf16x16 af = frag_a16(X + (size_t)arow * IND_ + kc + khalf * 16);
#pragma unroll
    for (int j = 0; j < 6; ++j) {
      const int col0 = (wave * 6 + j) * 16;
      bf16x16 bf = frag_b16(W + (size_t)(col0 + n) * IND_ + kc + khalf * 16);
      acc[j] = wmma_bf16(af, bf, acc[j]);
    }
  }
#pragma unroll
  for (int j = 0; j < 6; ++j) {
    const int col0 = (wave * 6 + j) * 16;
    const float bias = bb[col0 + n];
#pragma unroll
    for (int i = 0; i < 8; ++i)
      rowbuf[(i + 8 * khalf) * HID_ + col0 + n] = acc[j][i] + bias;
  }
  __syncthreads();

  // LayerNorm(768) + L2-normalize(768): wave handles rows 2w, 2w+1.
#pragma unroll
  for (int rr = 0; rr < 2; ++rr) {
    const int r = wave * 2 + rr;
    float s = 0.f, sq = 0.f;
    for (int c = lane; c < HID_; c += 32) {
      float v = rowbuf[r * HID_ + c];
      s += v; sq += v * v;
    }
    s = redadd32(s); sq = redadd32(sq);
    const float mean = s * (1.f / HID_);
    const float var  = sq * (1.f / HID_) - mean * mean;
    const float rs   = rsqrtf(var + 1e-5f);
    float nq = 0.f;
    for (int c = lane; c < HID_; c += 32) {
      float y = (rowbuf[r * HID_ + c] - mean) * rs * g[c] + be[c];
      rowbuf[r * HID_ + c] = y;
      nq += y * y;
    }
    nq = redadd32(nq);
    const float inv = 1.f / fmaxf(sqrtf(nq), 1e-12f);
    for (int c = lane; c < HID_; c += 32) {
      const float y = rowbuf[r * HID_ + c] * inv;
      Y[(size_t)(row0 + r) * HID_ + c]  = y;
      Yh[(size_t)(row0 + r) * HID_ + c] = (__bf16)y;
    }
  }
}

// ---------------------------------------------------------------------------
// Kernel 2: per-(b, head, direction) fused attention.
//   Phase A: K = X_kv @ Wk^T + bk  -> LDS (row-major [key][d], bf16)
//            V = X_kv @ Wv^T + bv  -> LDS (transposed  [d][key], bf16)
//   Phase B: per wave, 16 query rows: Q via WMMA, flash softmax over key
//            blocks of 32, O += P@V via WMMA, out-proj + bias + residual.
// Writes pre-LN concat directly into d_out [B,H,S,128].
// ---------------------------------------------------------------------------
__global__ __launch_bounds__(256) void attn_kernel(
    const float* __restrict__ imgN, const float* __restrict__ txtN,
    const __bf16* __restrict__ imgNh, const __bf16* __restrict__ txtNh,
    const __bf16* __restrict__ i2t_in_wh, const float* __restrict__ i2t_in_b,
    const __bf16* __restrict__ i2t_out_wh, const float* __restrict__ i2t_out_b,
    const __bf16* __restrict__ t2i_in_wh, const float* __restrict__ t2i_in_b,
    const __bf16* __restrict__ t2i_out_wh, const float* __restrict__ t2i_out_b,
    const unsigned char* __restrict__ mask, float* __restrict__ out) {
  __shared__ __bf16 k_sm[S_ * HD_];       // 64 KB, [key][d]
  __shared__ __bf16 vT_sm[HD_ * S_];      // 64 KB, [d][key]
  __shared__ __bf16 scr_sm[8][16 * HD_];  // 16 KB, per-wave relayout scratch

  const int tid = threadIdx.x, wave = tid >> 5, lane = tid & 31;
  const int n = lane & 15, khalf = lane >> 4;
  const int dir = blockIdx.x & 1;
  const int bh = blockIdx.x >> 1;
  const int b = bh / H_, h = bh % H_;

  const float*  qres = (dir ? txtN  : imgN)  + (size_t)b * S_ * HID_ + h * HD_;
  const __bf16* qh   = (dir ? txtNh : imgNh) + (size_t)b * S_ * HID_ + h * HD_;
  const __bf16* kvh  = (dir ? imgNh : txtNh) + (size_t)b * S_ * HID_ + h * HD_;
  const __bf16* in_wh  = (dir ? t2i_in_wh  : i2t_in_wh)  + (size_t)h * 192 * HD_;
  const float*  in_b   = (dir ? t2i_in_b   : i2t_in_b)   + (size_t)h * 192;
  const __bf16* out_wh = (dir ? t2i_out_wh : i2t_out_wh) + (size_t)h * HD_ * HD_;
  const float*  out_b  = (dir ? t2i_out_b  : i2t_out_b)  + (size_t)h * HD_;
  const unsigned char* mrow = mask + (size_t)b * S_;

  // ---- Phase A: K and V into LDS -----------------------------------------
  for (int job = wave; job < 256; job += 8) {
    const int sel = job >> 7;            // 0 = K, 1 = V
    const int rem = job & 127;
    const int mt = rem >> 2, nc = rem & 3;
    const int woff = 64 + sel * 64;      // Wk rows 64..127, Wv rows 128..191
    f32x8 acc = {};
    const int arow = mt * 16 + n;
#pragma unroll
    for (int kc = 0; kc < HD_; kc += 32) {
      bf16x16 af = frag_a16(kvh + (size_t)arow * HID_ + kc + khalf * 16);
      bf16x16 bf = frag_b16(in_wh + (size_t)(woff + nc * 16 + n) * HD_ + kc + khalf * 16);
      acc = wmma_bf16(af, bf, acc);
    }
    const float bias = in_b[woff + nc * 16 + n];
    if (sel == 0) {
#pragma unroll
      for (int i = 0; i < 8; ++i)
        k_sm[(size_t)(mt * 16 + i + 8 * khalf) * HD_ + nc * 16 + n] =
            (__bf16)(acc[i] + bias);
    } else {
#pragma unroll
      for (int i = 0; i < 8; ++i)
        vT_sm[(size_t)(nc * 16 + n) * S_ + mt * 16 + i + 8 * khalf] =
            (__bf16)(acc[i] + bias);
    }
  }
  __syncthreads();

  // ---- Phase B: flash attention, 16 query rows per wave per pass ---------
  __bf16* scr = scr_sm[wave];
  for (int pass = 0; pass < 4; ++pass) {
    const int qrow0 = (pass * 8 + wave) * 16;

    // Q tile: GEMM, then C-layout -> A-layout through LDS scratch.
    bf16x16 aq[2];
#pragma unroll
    for (int kc = 0; kc < 2; ++kc)
      aq[kc] = frag_a16(qh + (size_t)(qrow0 + n) * HID_ + kc * 32 + khalf * 16);
#pragma unroll
    for (int nc = 0; nc < 4; ++nc) {
      f32x8 acc = {};
#pragma unroll
      for (int kc = 0; kc < 2; ++kc) {
        bf16x16 bf = frag_b16(in_wh + (size_t)(nc * 16 + n) * HD_ + kc * 32 + khalf * 16);
        acc = wmma_bf16(aq[kc], bf, acc);
      }
      const float bias = in_b[nc * 16 + n];
#pragma unroll
      for (int i = 0; i < 8; ++i)
        scr[(i + 8 * khalf) * HD_ + nc * 16 + n] = (__bf16)(acc[i] + bias);
    }
    bf16x16 qa[2];
#pragma unroll
    for (int kc = 0; kc < 2; ++kc)
      qa[kc] = frag_a16(scr + n * HD_ + kc * 32 + khalf * 16);

    float mrun[8], lrun[8];
    f32x8 o[4];
#pragma unroll
    for (int i = 0; i < 8; ++i) { mrun[i] = -1e30f; lrun[i] = 0.f; }
#pragma unroll
    for (int nc = 0; nc < 4; ++nc) o[nc] = {};

    for (int kb = 0; kb < S_; kb += 32) {
      f32x8 s0 = {}, s1 = {};
#pragma unroll
      for (int kc = 0; kc < 2; ++kc) {
        s0 = wmma_bf16(qa[kc],
                       frag_b16(k_sm + (size_t)(kb + n) * HD_ + kc * 32 + khalf * 16), s0);
        s1 = wmma_bf16(qa[kc],
                       frag_b16(k_sm + (size_t)(kb + 16 + n) * HD_ + kc * 32 + khalf * 16), s1);
      }
      const float mk0 = mrow[kb + n]      ? 0.f : -1e30f;
      const float mk1 = mrow[kb + 16 + n] ? 0.f : -1e30f;
      float p0v[8], p1v[8];
#pragma unroll
      for (int i = 0; i < 8; ++i) {
        const float a0 = s0[i] * 0.125f + mk0;
        const float a1 = s1[i] * 0.125f + mk1;
        const float rm = redmax16(fmaxf(a0, a1));
        const float mnew = fmaxf(mrun[i], rm);
        const float alpha = __expf(mrun[i] - mnew);
        p0v[i] = __expf(a0 - mnew);
        p1v[i] = __expf(a1 - mnew);
        const float rs = redadd16(p0v[i] + p1v[i]);
        lrun[i] = lrun[i] * alpha + rs;
        mrun[i] = mnew;
#pragma unroll
        for (int nc = 0; nc < 4; ++nc) o[nc][i] *= alpha;
      }
      // P (16x32) -> bf16 LDS -> A-fragment.
#pragma unroll
      for (int i = 0; i < 8; ++i) {
        scr[(i + 8 * khalf) * 32 + n]      = (__bf16)p0v[i];
        scr[(i + 8 * khalf) * 32 + 16 + n] = (__bf16)p1v[i];
      }
      bf16x16 pa = frag_a16(scr + n * 32 + khalf * 16);
#pragma unroll
      for (int nc = 0; nc < 4; ++nc)
        o[nc] = wmma_bf16(pa,
                          frag_b16(vT_sm + (size_t)(nc * 16 + n) * S_ + kb + khalf * 16),
                          o[nc]);
    }

    // Normalize, out-projection, bias + residual, store pre-LN concat.
    float linv[8];
#pragma unroll
    for (int i = 0; i < 8; ++i) linv[i] = 1.f / lrun[i];
#pragma unroll
    for (int nc = 0; nc < 4; ++nc)
#pragma unroll
      for (int i = 0; i < 8; ++i)
        scr[(i + 8 * khalf) * HD_ + nc * 16 + n] = (__bf16)(o[nc][i] * linv[i]);
    bf16x16 ca[2];
#pragma unroll
    for (int kc = 0; kc < 2; ++kc)
      ca[kc] = frag_a16(scr + n * HD_ + kc * 32 + khalf * 16);
#pragma unroll
    for (int nc = 0; nc < 4; ++nc) {
      f32x8 acc = {};
#pragma unroll
      for (int kc = 0; kc < 2; ++kc) {
        bf16x16 bf = frag_b16(out_wh + (size_t)(nc * 16 + n) * HD_ + kc * 32 + khalf * 16);
        acc = wmma_bf16(ca[kc], bf, acc);
      }
      const float bias = out_b[nc * 16 + n];
#pragma unroll
      for (int i = 0; i < 8; ++i) {
        const int row = qrow0 + i + 8 * khalf;
        const float res = qres[(size_t)row * HID_ + nc * 16 + n];
        out[((size_t)(b * H_ + h) * S_ + row) * 128 + dir * 64 + nc * 16 + n] =
            acc[i] + bias + res;
      }
    }
  }
}

// ---------------------------------------------------------------------------
// Kernel 3: in-place LayerNorm over the 128-wide concat with per-head g/b.
// ---------------------------------------------------------------------------
__global__ __launch_bounds__(256) void hn_ln_kernel(
    float* __restrict__ out, const float* __restrict__ hn_g,
    const float* __restrict__ hn_b) {
  const int wave = threadIdx.x >> 5, lane = threadIdx.x & 31;
  const size_t row = (size_t)blockIdx.x * 8 + wave;
  const int h = (int)((row >> 9) % H_);  // row = (b*H + h)*S + s
  float4* p = (float4*)(out + row * 128);
  float4 x = p[lane];
  float s  = x.x + x.y + x.z + x.w;
  float sq = x.x * x.x + x.y * x.y + x.z * x.z + x.w * x.w;
  s = redadd32(s); sq = redadd32(sq);
  const float mean = s * (1.f / 128.f);
  const float var  = sq * (1.f / 128.f) - mean * mean;
  const float rs   = rsqrtf(var + 1e-5f);
  const float4 g  = ((const float4*)(hn_g + (size_t)h * 128))[lane];
  const float4 bb = ((const float4*)(hn_b + (size_t)h * 128))[lane];
  float4 y;
  y.x = (x.x - mean) * rs * g.x + bb.x;
  y.y = (x.y - mean) * rs * g.y + bb.y;
  y.z = (x.z - mean) * rs * g.z + bb.z;
  y.w = (x.w - mean) * rs * g.w + bb.w;
  p[lane] = y;
}

// ---------------------------------------------------------------------------
extern "C" void kernel_launch(void* const* d_in, const int* in_sizes, int n_in,
                              void* d_out, int out_size, void* d_ws,
                              size_t ws_size, hipStream_t stream) {
  const float* image    = (const float*)d_in[0];
  const float* text     = (const float*)d_in[1];
  const float* img_W    = (const float*)d_in[2];
  const float* img_b    = (const float*)d_in[3];
  const float* img_g    = (const float*)d_in[4];
  const float* img_beta = (const float*)d_in[5];
  const float* txt_W    = (const float*)d_in[6];
  const float* txt_b    = (const float*)d_in[7];
  const float* txt_g    = (const float*)d_in[8];
  const float* txt_beta = (const float*)d_in[9];
  const float* i2t_in_w  = (const float*)d_in[10];
  const float* i2t_in_b  = (const float*)d_in[11];
  const float* i2t_out_w = (const float*)d_in[12];
  const float* i2t_out_b = (const float*)d_in[13];
  const float* t2i_in_w  = (const float*)d_in[14];
  const float* t2i_in_b  = (const float*)d_in[15];
  const float* t2i_out_w = (const float*)d_in[16];
  const float* t2i_out_b = (const float*)d_in[17];
  const float* hn_g = (const float*)d_in[18];
  const float* hn_b = (const float*)d_in[19];
  const unsigned char* amask = (const unsigned char*)d_in[20];

  const size_t NX   = (size_t)B_ * S_ * HID_;   // 12,582,912 (normalized feats)
  const size_t NXin = (size_t)B_ * S_ * IND_;   //  8,388,608 (raw inputs)
  const size_t NWp  = (size_t)HID_ * IND_;      //    393,216 (proj weights)
  const size_t NWin = (size_t)H_ * 192 * HD_;   //    147,456 (QKV weights)
  const size_t NWo  = (size_t)H_ * HD_ * HD_;   //     49,152 (out-proj weights)

  float* ws   = (float*)d_ws;
  float* imgN = ws;                 // f32 normalized features
  float* txtN = imgN + NX;
  __bf16* hp       = (__bf16*)(txtN + NX);
  __bf16* imgNh    = hp;            hp += NX;
  __bf16* txtNh    = hp;            hp += NX;
  __bf16* imgXh    = hp;            hp += NXin;
  __bf16* txtXh    = hp;            hp += NXin;
  __bf16* imgWh    = hp;            hp += NWp;
  __bf16* txtWh    = hp;            hp += NWp;
  __bf16* i2t_iwh  = hp;            hp += NWin;
  __bf16* t2i_iwh  = hp;            hp += NWin;
  __bf16* i2t_owh  = hp;            hp += NWo;
  __bf16* t2i_owh  = hp;            hp += NWo;
  float* out = (float*)d_out;       // [B,H,S,128] f32

  // One-shot bf16 pre-conversions (inputs + all GEMM weights).
  cvt_bf16_kernel<<<NXin / 1024, 256, 0, stream>>>(image, imgXh);
  cvt_bf16_kernel<<<NXin / 1024, 256, 0, stream>>>(text, txtXh);
  cvt_bf16_kernel<<<NWp  / 1024, 256, 0, stream>>>(img_W, imgWh);
  cvt_bf16_kernel<<<NWp  / 1024, 256, 0, stream>>>(txt_W, txtWh);
  cvt_bf16_kernel<<<NWin / 1024, 256, 0, stream>>>(i2t_in_w, i2t_iwh);
  cvt_bf16_kernel<<<NWin / 1024, 256, 0, stream>>>(t2i_in_w, t2i_iwh);
  cvt_bf16_kernel<<<NWo  / 1024, 256, 0, stream>>>(i2t_out_w, i2t_owh);
  cvt_bf16_kernel<<<NWo  / 1024, 256, 0, stream>>>(t2i_out_w, t2i_owh);

  proj_norm_kernel<<<dim3(B_ * S_ / 16, 2), 256, 0, stream>>>(
      imgXh, txtXh, imgWh, img_b, img_g, img_beta, txtWh, txt_b, txt_g,
      txt_beta, imgN, txtN, imgNh, txtNh);

  attn_kernel<<<B_ * H_ * 2, 256, 0, stream>>>(
      imgN, txtN, imgNh, txtNh, i2t_iwh, i2t_in_b, i2t_owh, i2t_out_b,
      t2i_iwh, t2i_in_b, t2i_owh, t2i_out_b, amask, out);

  hn_ln_kernel<<<B_ * H_ * S_ / 8, 256, 0, stream>>>(out, hn_g, hn_b);
}
